// MultiHeadAttention_27831388078819
// MI455X (gfx1250) — compile-verified, matrix operand order checked by reading, not run on hardware
//
#include <hip/hip_runtime.h>
#include <hip/hip_bf16.h>

typedef unsigned short u16;
typedef unsigned int   u32;

typedef __attribute__((ext_vector_type(16))) __bf16 v16bf;
typedef __attribute__((ext_vector_type(8)))  float  v8f;
typedef __attribute__((ext_vector_type(8)))  u32    v8u;

// Problem constants (from reference)
#define EDIM 1024
#define SEQ  2048
#define NB   2
#define NH   16
#define HD   64
#define MTOT (NB * SEQ)            // 4096 rows of activations

// ---------------------------------------------------------------------------
// helpers
// ---------------------------------------------------------------------------
__device__ __forceinline__ u16 f2bf(float f) {
  u32 x = __float_as_uint(f);
  x += 0x7FFFu + ((x >> 16) & 1u);   // round-to-nearest-even
  return (u16)(x >> 16);
}

__device__ __forceinline__ v8f vzero8() {
  v8f z;
#pragma unroll
  for (int i = 0; i < 8; ++i) z[i] = 0.0f;
  return z;
}

__device__ __forceinline__ v8f wmma_bf16(v16bf a, v16bf b, v8f c) {
  // D = A(16x32 bf16) * B(32x16 bf16) + C(16x16 f32)
  return __builtin_amdgcn_wmma_f32_16x16x32_bf16(
      false, a, false, b, (short)0, c, false, false);
}

// Gather a 16x32 bf16 A-fragment (or B-fragment of the transposed operand)
// straight from row-major global memory.
// Per-lane layout (ISA 7.12.2, 16-bit A 16x32):
//   row  = row0 + (lane & 15)
//   elem e (0..15): k = (e/8)*16 + (lane/16)*8 + (e%8)  -> contiguous pairs
__device__ __forceinline__ v16bf load_frag_g(const u16* __restrict__ p,
                                             int row0, int ld, int k0,
                                             int lane) {
  int r  = row0 + (lane & 15);
  int hk = (lane >> 4) << 3;                       // 0 or 8
  const u32* base = (const u32*)(p + (size_t)r * ld + k0);
  v8u u;
#pragma unroll
  for (int j = 0; j < 8; ++j) {
    int k = ((j >> 2) << 4) + hk + ((j & 3) << 1); // even -> dword aligned
    u[j] = base[k >> 1];
  }
  return __builtin_bit_cast(v16bf, u);
}

// ---------------------------------------------------------------------------
// fp32 -> bf16 conversion (grid-stride)
// ---------------------------------------------------------------------------
__global__ __launch_bounds__(256) void cvt_f32_bf16(const float* __restrict__ in,
                                                    u16* __restrict__ out,
                                                    int n) {
  for (int i = blockIdx.x * blockDim.x + threadIdx.x; i < n;
       i += gridDim.x * blockDim.x)
    out[i] = f2bf(in[i]);
}

// ---------------------------------------------------------------------------
// Projection GEMM:  out = A(4096x1024) @ W^T(1024x1024)  -> bf16, head-major
//   vtrans==0 : out[b][h][s][d]      (Q,K)
//   vtrans==1 : out[b][h][d][s]      (V, pre-transposed for context GEMM)
// Block = 256 thr (8 waves), per-wave 32x64 tile, block tile 64x256.
// ---------------------------------------------------------------------------
__global__ __launch_bounds__(256) void gemm_qkv(const u16* __restrict__ A,
                                                const u16* __restrict__ W,
                                                u16* __restrict__ out,
                                                int vtrans) {
  int lane = threadIdx.x & 31;
  int wave = threadIdx.x >> 5;
  int m0 = blockIdx.y * 64 + (wave & 1) * 32;
  int n0 = blockIdx.x * 256 + (wave >> 1) * 64;

  v8f acc[2][4];
#pragma unroll
  for (int i = 0; i < 2; ++i)
#pragma unroll
    for (int t = 0; t < 4; ++t) acc[i][t] = vzero8();

  for (int k0 = 0; k0 < EDIM; k0 += 32) {
    v16bf a0 = load_frag_g(A, m0,      EDIM, k0, lane);
    v16bf a1 = load_frag_g(A, m0 + 16, EDIM, k0, lane);
#pragma unroll
    for (int t = 0; t < 4; ++t) {
      v16bf b = load_frag_g(W, n0 + t * 16, EDIM, k0, lane);  // B[k,n]=W[n,k]
      acc[0][t] = wmma_bf16(a0, b, acc[0][t]);
      acc[1][t] = wmma_bf16(a1, b, acc[1][t]);
    }
  }

  int ncol = lane & 15, hi = lane >> 4;
#pragma unroll
  for (int i = 0; i < 2; ++i)
#pragma unroll
    for (int t = 0; t < 4; ++t)
#pragma unroll
      for (int j = 0; j < 8; ++j) {
        int m   = m0 + i * 16 + j + 8 * hi;
        int col = n0 + t * 16 + ncol;
        int bb = m >> 11, s = m & (SEQ - 1);
        int hh = col >> 6, dd = col & (HD - 1);
        size_t idx = vtrans
            ? ((((size_t)bb * NH + hh) * HD + dd) * SEQ + s)
            : ((((size_t)bb * NH + hh) * SEQ + s) * HD + dd);
        out[idx] = f2bf(acc[i][t][j]);
      }
}

// ---------------------------------------------------------------------------
// Output projection: out = AO(4096x1024 bf16) @ Wo^T + bo  -> fp32
// ---------------------------------------------------------------------------
__global__ __launch_bounds__(256) void gemm_out(const u16* __restrict__ A,
                                                const u16* __restrict__ W,
                                                const float* __restrict__ bias,
                                                float* __restrict__ out) {
  int lane = threadIdx.x & 31;
  int wave = threadIdx.x >> 5;
  int m0 = blockIdx.y * 64 + (wave & 1) * 32;
  int n0 = blockIdx.x * 256 + (wave >> 1) * 64;

  v8f acc[2][4];
#pragma unroll
  for (int i = 0; i < 2; ++i)
#pragma unroll
    for (int t = 0; t < 4; ++t) acc[i][t] = vzero8();

  for (int k0 = 0; k0 < EDIM; k0 += 32) {
    v16bf a0 = load_frag_g(A, m0,      EDIM, k0, lane);
    v16bf a1 = load_frag_g(A, m0 + 16, EDIM, k0, lane);
#pragma unroll
    for (int t = 0; t < 4; ++t) {
      v16bf b = load_frag_g(W, n0 + t * 16, EDIM, k0, lane);
      acc[0][t] = wmma_bf16(a0, b, acc[0][t]);
      acc[1][t] = wmma_bf16(a1, b, acc[1][t]);
    }
  }

  int ncol = lane & 15, hi = lane >> 4;
#pragma unroll
  for (int i = 0; i < 2; ++i)
#pragma unroll
    for (int t = 0; t < 4; ++t)
#pragma unroll
      for (int j = 0; j < 8; ++j) {
        int m   = m0 + i * 16 + j + 8 * hi;
        int col = n0 + t * 16 + ncol;
        out[(size_t)m * EDIM + col] = acc[i][t][j] + bias[col];
      }
}

// ---------------------------------------------------------------------------
// Flash-style attention. 1 wave = 16 query rows; block = 8 waves = 128 rows.
// Streams keys in 32-wide steps with online softmax; scores never hit HBM.
// Q,K: [B,H,S,64] bf16;  Vt: [B,H,64,S] bf16;  mask: [B,1,S,S] i32
// AO: [B,S,1024] bf16 (col = h*64 + d), feeds the output projection.
// ---------------------------------------------------------------------------
__global__ __launch_bounds__(256) void attn_kernel(const u16* __restrict__ Q,
                                                   const u16* __restrict__ K,
                                                   const u16* __restrict__ Vt,
                                                   const int* __restrict__ mask,
                                                   u16* __restrict__ AO) {
  __shared__ u16 lds_p[8][16][32];   // per-wave probability tile (C->A relayout)

  int lane = threadIdx.x & 31;
  int wave = threadIdx.x >> 5;
  const int qtiles = SEQ / 128;      // 16
  int bh   = blockIdx.x / qtiles;
  int qblk = blockIdx.x % qtiles;
  int b = bh / NH, h = bh % NH;
  int q0 = qblk * 128 + wave * 16;

  const u16* Qb = Q  + ((size_t)(b * NH + h) * SEQ) * HD;
  const u16* Kb = K  + ((size_t)(b * NH + h) * SEQ) * HD;
  const u16* Vb = Vt + ((size_t)(b * NH + h) * HD) * SEQ;
  const int* Mb = mask + (size_t)b * SEQ * SEQ;

  // Q fragments stay resident for the whole key loop
  v16bf qa0 = load_frag_g(Qb, q0, HD, 0,  lane);
  v16bf qa1 = load_frag_g(Qb, q0, HD, 32, lane);

  v8f o[4];
#pragma unroll
  for (int t = 0; t < 4; ++t) o[t] = vzero8();
  float mrow[8], lrow[8];
#pragma unroll
  for (int j = 0; j < 8; ++j) { mrow[j] = -1e30f; lrow[j] = 0.0f; }

  int ncol = lane & 15, hi = lane >> 4;
  const float scale = 0.03125f;           // 1/sqrt(1024), per reference

  for (int key0 = 0; key0 < SEQ; key0 += 32) {
    // ---- scores: two 16x16 tiles, K-dim 64 in two WMMA steps each
    v8f sc[2];
#pragma unroll
    for (int t = 0; t < 2; ++t) {
      v16bf kb0 = load_frag_g(Kb, key0 + t * 16, HD, 0,  lane);
      v16bf kb1 = load_frag_g(Kb, key0 + t * 16, HD, 32, lane);
      v8f c = vzero8();
      c = wmma_bf16(qa0, kb0, c);
      c = wmma_bf16(qa1, kb1, c);
      sc[t] = c;
    }

    // ---- mask + scale + online softmax (rows span the 16-lane column group)
    float corr[8];
#pragma unroll
    for (int j = 0; j < 8; ++j) {
      int qrow = q0 + j + 8 * hi;
      const int* mp = Mb + (size_t)qrow * SEQ + key0;
      // reference: masked -> -1e20 BEFORE the /32 scaling => -3.125e18
      float s0 = mp[ncol]      ? sc[0][j] * scale : -3.125e18f;
      float s1 = mp[16 + ncol] ? sc[1][j] * scale : -3.125e18f;
      float tm = fmaxf(s0, s1);
#pragma unroll
      for (int off = 1; off < 16; off <<= 1)
        tm = fmaxf(tm, __shfl_xor(tm, off, 32));
      float nm = fmaxf(mrow[j], tm);
      corr[j]  = __expf(mrow[j] - nm);
      mrow[j]  = nm;
      float p0 = __expf(s0 - nm);
      float p1 = __expf(s1 - nm);
      float rs = p0 + p1;
#pragma unroll
      for (int off = 1; off < 16; off <<= 1)
        rs += __shfl_xor(rs, off, 32);
      lrow[j] = lrow[j] * corr[j] + rs;
      // stash probabilities (C layout) in LDS, row-major [16][32]
      int r = j + 8 * hi;
      lds_p[wave][r][ncol]      = f2bf(p0);
      lds_p[wave][r][16 + ncol] = f2bf(p1);
    }

    // same-wave LDS RAW: wait for ds stores before re-reading in A layout
    asm volatile("s_wait_dscnt 0" ::: "memory");

    v8u pu;
    {
      int r  = ncol;
      int hk = hi << 3;
#pragma unroll
      for (int j = 0; j < 8; ++j) {
        int k = ((j >> 2) << 4) + hk + ((j & 3) << 1);
        pu[j] = *(const u32*)&lds_p[wave][r][k];
      }
    }
    v16bf pa = __builtin_bit_cast(v16bf, pu);

    // ---- context: O(16x64) += P(16x32) @ V(32x64), with max-correction
#pragma unroll
    for (int t = 0; t < 4; ++t) {
      v16bf vb = load_frag_g(Vb, t * 16, SEQ, key0, lane);  // Vt rows contiguous
#pragma unroll
      for (int j = 0; j < 8; ++j) o[t][j] *= corr[j];
      o[t] = wmma_bf16(pa, vb, o[t]);
    }
  }

  // ---- epilogue: normalize and emit AO (bf16, col = h*64+d)
#pragma unroll
  for (int j = 0; j < 8; ++j) lrow[j] = 1.0f / lrow[j];
#pragma unroll
  for (int t = 0; t < 4; ++t)
#pragma unroll
    for (int j = 0; j < 8; ++j) {
      int qrow = q0 + j + 8 * hi;
      size_t idx = ((size_t)b * SEQ + qrow) * EDIM + h * HD + t * 16 + ncol;
      AO[idx] = f2bf(o[t][j] * lrow[j]);
    }
}

// ---------------------------------------------------------------------------
// launch
// ---------------------------------------------------------------------------
extern "C" void kernel_launch(void* const* d_in, const int* in_sizes, int n_in,
                              void* d_out, int out_size, void* d_ws, size_t ws_size,
                              hipStream_t stream) {
  const float* value = (const float*)d_in[0];
  const float* keyin = (const float*)d_in[1];
  const float* query = (const float*)d_in[2];
  const int*   mask  = (const int*)  d_in[3];
  const float* Wv    = (const float*)d_in[4];
  const float* Wk    = (const float*)d_in[5];
  const float* Wq    = (const float*)d_in[6];
  const float* Wo    = (const float*)d_in[7];
  const float* bo    = (const float*)d_in[8];
  float* out = (float*)d_out;

  const size_t MS = (size_t)MTOT * EDIM;   // 4,194,304 elems
  const size_t WW = (size_t)EDIM * EDIM;   // 1,048,576 elems

  u16* ws    = (u16*)d_ws;
  u16* v_bf  = ws;               ws += MS;
  u16* k_bf  = ws;               ws += MS;
  u16* q_bf  = ws;               ws += MS;
  u16* Wv_bf = ws;               ws += WW;
  u16* Wk_bf = ws;               ws += WW;
  u16* Wq_bf = ws;               ws += WW;
  u16* Wo_bf = ws;               ws += WW;
  u16* q_pr  = ws;               ws += MS;   // [B,H,S,64]
  u16* k_pr  = ws;               ws += MS;   // [B,H,S,64]
  u16* v_prT = ws;               ws += MS;   // [B,H,64,S]
  u16* AO    = ws;               ws += MS;   // [B,S,1024]

  // fp32 -> bf16 staging
  cvt_f32_bf16<<<2048, 256, 0, stream>>>(value, v_bf, (int)MS);
  cvt_f32_bf16<<<2048, 256, 0, stream>>>(keyin, k_bf, (int)MS);
  cvt_f32_bf16<<<2048, 256, 0, stream>>>(query, q_bf, (int)MS);
  cvt_f32_bf16<<<1024, 256, 0, stream>>>(Wv, Wv_bf, (int)WW);
  cvt_f32_bf16<<<1024, 256, 0, stream>>>(Wk, Wk_bf, (int)WW);
  cvt_f32_bf16<<<1024, 256, 0, stream>>>(Wq, Wq_bf, (int)WW);
  cvt_f32_bf16<<<1024, 256, 0, stream>>>(Wo, Wo_bf, (int)WW);

  // QKV projections (V emitted pre-transposed per head)
  dim3 gg(EDIM / 256, MTOT / 64);
  gemm_qkv<<<gg, 256, 0, stream>>>(q_bf, Wq_bf, q_pr, 0);
  gemm_qkv<<<gg, 256, 0, stream>>>(k_bf, Wk_bf, k_pr, 0);
  gemm_qkv<<<gg, 256, 0, stream>>>(v_bf, Wv_bf, v_prT, 1);

  // streaming-softmax attention
  attn_kernel<<<NB * NH * (SEQ / 128), 256, 0, stream>>>(q_pr, k_pr, v_prT,
                                                         mask, AO);

  // output projection + bias (fp32 out)
  gemm_out<<<gg, 256, 0, stream>>>(AO, Wo_bf, bo, out);
}
